// ElmanRNN_24103356465684
// MI455X (gfx1250) — matrix operand order repaired and checked by reading, not compiled, not run on hardware
//
#include <hip/hip_runtime.h>
#include <math.h>

// Problem constants (from reference): B=64, S=2048, I=256, H=512
#define RNN_B 64
#define RNN_S 2048
#define RNN_I 256
#define RNN_H 512

// gfx1250 has the bf16 cvt ISA but this clang doesn't declare the builtin;
// use inline asm. Set to 0 to fall back to manual packing if asm fails.
#define USE_ASM_CVT_PK_BF16 1

typedef __attribute__((ext_vector_type(16))) __bf16       v16bf;
typedef __attribute__((ext_vector_type(8)))  float        v8f;
typedef __attribute__((ext_vector_type(4)))  unsigned int uint4v;
typedef __attribute__((ext_vector_type(2)))  unsigned int uint2v;

union FragU { v16bf v; uint4v q[2]; };

// Pack two f32 -> two bf16 in one dword: lo in [15:0], hi in [31:16].
__device__ inline unsigned int pack2bf(float lo, float hi) {
#if USE_ASM_CVT_PK_BF16
    unsigned int r;
    asm("v_cvt_pk_bf16_f32 %0, %1, %2" : "=v"(r) : "v"(lo), "v"(hi));
    return r;
#else
    unsigned int a = __float_as_uint(lo) + 0x8000u;   // round-half-up (cheap)
    unsigned int b = __float_as_uint(hi) + 0x8000u;
    return (a >> 16) | (b & 0xffff0000u);
#endif
}

__device__ inline unsigned short f2bf(float f) {
    return (unsigned short)pack2bf(f, f);
}

__device__ inline uint2v pack4bf(float4 v) {
    uint2v r;
    r.x = pack2bf(v.x, v.y);
    r.y = pack2bf(v.z, v.w);
    return r;
}

// Fast tanh on the HW transcendental path: tanh(x) = 1 - 2/(exp2(2*log2e*x)+1).
// Saturates correctly: x -> +inf => 1, x -> -inf => -1.
__device__ inline float tanh_fast(float x) {
    float t = __builtin_amdgcn_exp2f(x * 2.88539008177792681f);  // exp(2x)
    return 1.0f - 2.0f * __builtin_amdgcn_rcpf(t + 1.0f);
}

// A-matrix fragment (16x32 bf16) from a row-major bf16 buffer (stride ldk elems).
// ISA layout: lanes 0-15 -> M=lane, K = {0..7, 16..23}; lanes 16-31 -> K = {8..15, 24..31}.
__device__ inline v16bf load_frag_a(const unsigned short* p, int ldk, int mBase, int k0) {
    int lane  = threadIdx.x & 31;
    int m     = mBase + (lane & 15);
    int kbase = k0 + ((lane >> 4) << 3);   // +0 or +8
    FragU f;
    f.q[0] = *reinterpret_cast<const uint4v*>(p + m * ldk + kbase);       // K kbase..+7
    f.q[1] = *reinterpret_cast<const uint4v*>(p + m * ldk + kbase + 16);  // K kbase+16..+23
    return f.v;
}

// B-matrix fragment (32x16 bf16). Source stores B^T row-major: p[n*ldk + k] = B[k][n].
// ISA layout: lanes 0-15 -> N=lane, K = {0..7, 16..23}; lanes 16-31 -> N=lane-16, K = {8..15, 24..31}.
__device__ inline v16bf load_frag_b(const unsigned short* p, int ldk, int nBase, int k0) {
    int lane = threadIdx.x & 31;
    int n    = nBase + (lane & 15);
    int k    = k0 + ((lane >> 4) << 3);    // +0 or +8
    FragU f;
    f.q[0] = *reinterpret_cast<const uint4v*>(p + n * ldk + k);
    f.q[1] = *reinterpret_cast<const uint4v*>(p + n * ldk + k + 16);
    return f.v;
}

// ---------------------------------------------------------------------------
// Kernel 1: x_proj[r, h] = sum_i X[r,i] * W_ih[h,i] + b_ih[h] + b_hh[h]
// r = b*S + s flattened; result written straight into d_out ([B,S,H] layout).
// 64x64 output tile per workgroup, 8 waves (4 M-tiles x 2 N-halves, 2 N-subtiles each).
// Staging: global_load_b128 (float4) -> v_cvt_pk_bf16_f32 -> ds_store_b64.
// ---------------------------------------------------------------------------
__global__ __launch_bounds__(256) void xproj_kernel(
    const float* __restrict__ x, const float* __restrict__ Wih,
    const float* __restrict__ bih, const float* __restrict__ bhh,
    float* __restrict__ out) {
    __shared__ __align__(16) unsigned short As[64 * 32];   // 4 KB
    __shared__ __align__(16) unsigned short Ws[64 * 32];   // 4 KB

    const int nTiles = RNN_H / 64;                         // 8
    const int tileN  = blockIdx.x % nTiles;
    const int tileM  = blockIdx.x / nTiles;
    const int row0   = tileM * 64;
    const int col0   = tileN * 64;
    const int tid    = threadIdx.x;
    const int lane   = tid & 31;
    const int wid    = tid >> 5;
    const int wm     = wid & 3;       // M-tile 0..3
    const int wn     = wid >> 2;      // N-half 0..1

    // This thread's (row, k-quad) within a staged 64x32 block.
    const int srow = tid >> 3;             // 0..31  (j adds +32)
    const int skq  = (tid & 7) << 2;       // 0,4,...,28

    v8f acc0 = {};
    v8f acc1 = {};

    for (int k0 = 0; k0 < RNN_I; k0 += 32) {
        // Prefetch next k-slab of the big streaming x read (near scope).
        if (k0 + 32 < RNN_I) {
            __builtin_prefetch(&x[(long)(row0 + srow) * RNN_I + k0 + 32 + skq], 0, 3);
        }
        __syncthreads();
        // Stage 64x32 blocks of X and W_ih as bf16: 512 float4 each, 2 per thread.
        #pragma unroll
        for (int j = 0; j < 2; ++j) {
            int r  = srow + 32 * j;
            float4 xa = *reinterpret_cast<const float4*>(
                            &x[(long)(row0 + r) * RNN_I + k0 + skq]);
            float4 wa = *reinterpret_cast<const float4*>(
                            &Wih[(col0 + r) * RNN_I + k0 + skq]);
            *reinterpret_cast<uint2v*>(&As[r * 32 + skq]) = pack4bf(xa);
            *reinterpret_cast<uint2v*>(&Ws[r * 32 + skq]) = pack4bf(wa);
        }
        __syncthreads();

        v16bf a  = load_frag_a(As, 32, wm * 16, 0);
        v16bf b0 = load_frag_b(Ws, 32, wn * 32 + 0, 0);
        v16bf b1 = load_frag_b(Ws, 32, wn * 32 + 16, 0);
        acc0 = __builtin_amdgcn_wmma_f32_16x16x32_bf16(false, a, false, b0,
                                                       (short)0, acc0, false, false);
        acc1 = __builtin_amdgcn_wmma_f32_16x16x32_bf16(false, a, false, b1,
                                                       (short)0, acc1, false, false);
    }

    // C layout: lane n = lane&15; rows = (lane>>4)*8 + v.
    const int n    = lane & 15;
    const int mrow = (lane >> 4) * 8;
    const int c0   = col0 + wn * 32 + n;
    const int c1   = c0 + 16;
    const float bias0 = bih[c0] + bhh[c0];
    const float bias1 = bih[c1] + bhh[c1];
    #pragma unroll
    for (int v = 0; v < 8; ++v) {
        long r = row0 + wm * 16 + mrow + v;
        out[r * RNN_H + c0] = acc0[v] + bias0;
        out[r * RNN_H + c1] = acc1[v] + bias1;
    }
}

// ---------------------------------------------------------------------------
// Init kernel: h0 (f32) -> bf16 ping buffer; reset grid-sync counter.
// ---------------------------------------------------------------------------
__global__ __launch_bounds__(256) void init_kernel(
    const float* __restrict__ h0, unsigned int* __restrict__ hA32,
    unsigned int* __restrict__ counter) {
    int i = blockIdx.x * 256 + threadIdx.x;   // 0 .. B*H/2 - 1
    float2 v = *reinterpret_cast<const float2*>(&h0[i * 2]);
    hA32[i] = pack2bf(v.x, v.y);
    if (i == 0) *counter = 0u;
}

// ---------------------------------------------------------------------------
// Kernel 2: persistent recurrence.  32 WGs, each owns a 16-column strip of H.
// Per step:  h_new = tanh(x_proj[s] + h_old @ W_hh^T), updated in-place in d_out,
// bf16 h ping-pongs through workspace (L2-resident), grid barrier between steps.
// 8 waves = 4 B-row-tiles x 2 K-halves (split-K reduced through LDS).
// ---------------------------------------------------------------------------
__global__ __launch_bounds__(256) void rnn_recurrence_kernel(
    const float* __restrict__ Whh, float* __restrict__ out,
    unsigned short* __restrict__ hA, unsigned short* __restrict__ hB,
    unsigned int* __restrict__ counter) {
    __shared__ __align__(16) unsigned short Wt[16 * RNN_H]; // strip of W_hh^T, 16 KB
    __shared__ __align__(16) float          scr[4 * 256];   // split-K partials, 4 KB

    const int tid  = threadIdx.x;
    const int lane = tid & 31;
    const int wid  = tid >> 5;
    const int wm   = wid & 3;          // B-row tile (16 rows)
    const int kh   = wid >> 2;         // K half: 0 -> K[0,256), 1 -> K[256,512)
    const int col0 = blockIdx.x * 16;  // this WG's H-strip
    const unsigned int nwg = gridDim.x;

    // Load this strip of W_hh (B^T layout: Wt[n*512 + k] = W_hh[col0+n][k]) once.
    // 2048 float4 total, 8 per thread; vectorized pack + b64 LDS stores.
    #pragma unroll
    for (int j = 0; j < 8; ++j) {
        int l4 = tid + 256 * j;              // 0..2047
        int n  = l4 >> 7;                    // 128 float4 per 512-wide row
        int kq = (l4 & 127) << 2;
        float4 wa = *reinterpret_cast<const float4*>(&Whh[(col0 + n) * RNN_H + kq]);
        *reinterpret_cast<uint2v*>(&Wt[n * RNN_H + kq]) = pack4bf(wa);
    }
    __syncthreads();

    for (int s = 0; s < RNN_S; ++s) {
        const unsigned short* hcur = (s & 1) ? hB : hA;
        unsigned short*       hnxt = (s & 1) ? hA : hB;

        // Split-K WMMA: this wave covers rows [wm*16, +16), K [kh*256, +256).
        v8f acc = {};
        #pragma unroll
        for (int kk = 0; kk < 8; ++kk) {
            int k0 = kh * 256 + kk * 32;
            v16bf a = load_frag_a(hcur, RNN_H, wm * 16, k0);  // from global (L2-hot)
            v16bf b = load_frag_b(Wt, RNN_H, 0, k0);          // from LDS
            acc = __builtin_amdgcn_wmma_f32_16x16x32_bf16(false, a, false, b,
                                                          (short)0, acc, false, false);
        }

        if (kh == 1) {
            #pragma unroll
            for (int v = 0; v < 8; ++v) scr[wm * 256 + lane * 8 + v] = acc[v];
        }
        __syncthreads();

        if (kh == 0) {
            const int n   = lane & 15;
            const int col = col0 + n;
            #pragma unroll
            for (int v = 0; v < 8; ++v) {
                float dot = acc[v] + scr[wm * 256 + lane * 8 + v];
                int   b   = wm * 16 + (lane >> 4) * 8 + v;      // batch row
                long  oix = ((long)b * RNN_S + s) * RNN_H + col;
                float hv  = tanh_fast(out[oix] + dot);          // out holds x_proj
                out[oix]       = hv;                            // in-place hidden out
                hnxt[b * RNN_H + col] = f2bf(hv);
            }
        }

        // ---- grid-wide barrier: all 32 strips must finish step s ----
        __threadfence();
        __syncthreads();
        if (tid == 0) {
            __hip_atomic_fetch_add(counter, 1u, __ATOMIC_RELEASE,
                                   __HIP_MEMORY_SCOPE_AGENT);
            unsigned int target = nwg * (unsigned int)(s + 1);
            while (__hip_atomic_load(counter, __ATOMIC_ACQUIRE,
                                     __HIP_MEMORY_SCOPE_AGENT) < target) {
                __builtin_amdgcn_s_sleep(2);
            }
        }
        __syncthreads();
    }
}

// ---------------------------------------------------------------------------
// kernel_launch
// d_in order: x_in[B,S,I], h0[B,H], W_ih[H,I], W_hh[H,H], b_ih[H], b_hh[H]
// d_out: [B,S,H] f32. Workspace: counter (256 B) + two bf16 h buffers (64 KB each).
// ---------------------------------------------------------------------------
extern "C" void kernel_launch(void* const* d_in, const int* in_sizes, int n_in,
                              void* d_out, int out_size, void* d_ws, size_t ws_size,
                              hipStream_t stream) {
    (void)in_sizes; (void)n_in; (void)out_size; (void)ws_size;
    const float* x   = (const float*)d_in[0];
    const float* h0  = (const float*)d_in[1];
    const float* Wih = (const float*)d_in[2];
    const float* Whh = (const float*)d_in[3];
    const float* bih = (const float*)d_in[4];
    const float* bhh = (const float*)d_in[5];
    float* out = (float*)d_out;

    char* ws = (char*)d_ws;
    unsigned int*   counter = (unsigned int*)ws;
    unsigned short* hA      = (unsigned short*)(ws + 256);
    unsigned short* hB      = hA + RNN_B * RNN_H;

    // Phase 0: init h ping buffer + barrier counter (deterministic per launch).
    init_kernel<<<(RNN_B * RNN_H / 2) / 256, 256, 0, stream>>>(
        h0, (unsigned int*)hA, counter);

    // Phase 1: big GEMM, x_proj (+biases) written straight into d_out.
    const int mTiles = (RNN_B * RNN_S) / 64;   // 2048
    const int nTiles = RNN_H / 64;             // 8
    xproj_kernel<<<mTiles * nTiles, 256, 0, stream>>>(x, Wih, bih, bhh, out);

    // Phase 2: persistent recurrence, one WG per 16-column strip of H.
    rnn_recurrence_kernel<<<RNN_H / 16, 256, 0, stream>>>(Whh, out, hA, hB, counter);
}
